// GCN_5686536700059
// MI455X (gfx1250) — matrix operand order repaired and checked by reading, not compiled, not run on hardware
//
#include <hip/hip_runtime.h>

// ---------------------------------------------------------------------------
// GCN (3x GCNConv + log_softmax) for MI455X / gfx1250, wave32, WMMA bf16.
//
// Sizes (reference): N=100000 nodes, E=1.6M edges, D=H=128, C=16.
// Dense GEMMs: ~6.9 GFLOP total -> WMMA bf16 (f32 accum), trivial runtime.
// Aggregation: t (51.2MB) + h (51.2MB) fit in the 192MB L2, so the random
// gather + f32-atomic scatter traffic is L2-resident; activations stay f32
// for atomics, with a compact bf16 shadow copy as the WMMA A operand.
// Weights are pre-packed *transposed* (Wt[n][k]) so every WMMA B fragment is
// two contiguous global_load_b128 per lane instead of 16 strided u16 loads.
// ---------------------------------------------------------------------------

typedef __attribute__((ext_vector_type(16))) __bf16 v16bf;
typedef __attribute__((ext_vector_type(8)))  __bf16 v8bf;
typedef __attribute__((ext_vector_type(8)))  float  v8f;

static __device__ __forceinline__ unsigned short f2bf(float f) {
    union { float f; unsigned u; } v; v.f = f;
    unsigned u = v.u;
    u += 0x7FFFu + ((u >> 16) & 1u);        // round-to-nearest-even
    return (unsigned short)(u >> 16);
}

// ---------------- degree / normalization ----------------

__global__ void k_deg_init(float* __restrict__ deg, int n) {
    int i = blockIdx.x * blockDim.x + threadIdx.x;
    if (i < n) deg[i] = 1.0f;               // self-loop contributes 1
}

__global__ void k_deg_edges(const int* __restrict__ dst, float* __restrict__ deg, int e) {
    int i = blockIdx.x * blockDim.x + threadIdx.x;
    if (i < e) atomicAdd(&deg[dst[i]], 1.0f);
}

__global__ void k_dinv(const float* __restrict__ deg, float* __restrict__ dinv, int n) {
    int i = blockIdx.x * blockDim.x + threadIdx.x;
    if (i < n) dinv[i] = rsqrtf(deg[i]);    // deg >= 1 always (self-loop)
}

// ---------------- f32 -> bf16 converts ----------------

// activations: plain elementwise
__global__ void k_f2bf(const float* __restrict__ in, unsigned short* __restrict__ out, int n) {
    int i = blockIdx.x * blockDim.x + threadIdx.x;
    if (i < n) out[i] = f2bf(in[i]);
}

// weights: convert + transpose. In: W[k][n] (K x Ncols f32). Out: Wt[n][k] bf16.
__global__ void k_packW(const float* __restrict__ W, unsigned short* __restrict__ Wt,
                        int K, int Ncols) {
    int i = blockIdx.x * blockDim.x + threadIdx.x;  // i = n*K + k
    if (i < K * Ncols) {
        const int n = i / K;
        const int k = i - n * K;
        Wt[i] = f2bf(W[(size_t)k * Ncols + n]);
    }
}

// ---------------- WMMA bf16 GEMM: C[nRows x Ncols] = A[nRows x K] @ B[K x Ncols]
// One wave computes one 16x16 C tile. A row-major bf16, B given transposed
// (Bt[n][k], n-major) bf16; C f32. Fragment layouts per CDNA5 ISA 7.12.2.

__global__ void k_gemm_bf16(const unsigned short* __restrict__ Abits,
                            const unsigned short* __restrict__ Btbits,
                            float* __restrict__ C,
                            int nRows, int K, int Ncols) {
    const int wid  = (blockIdx.x * blockDim.x + threadIdx.x) >> 5;
    const int lane = threadIdx.x & 31;
    const int tilesN   = Ncols >> 4;
    const int tileRow  = wid / tilesN;
    const int tileCol  = wid - tileRow * tilesN;
    const int rowTiles = (nRows + 15) >> 4;
    if (tileRow >= rowTiles) return;        // wave-uniform: EXEC stays all-ones

    const int rowBase = tileRow << 4;
    const int colBase = tileCol << 4;
    const int half    = lane >> 4;          // 0: low K-half lanes, 1: high
    const int l15     = lane & 15;

    int aRow = rowBase + l15;
    if (aRow >= nRows) aRow = nRows - 1;    // clamped duplicate load for tail
    const __bf16* A  = (const __bf16*)Abits;
    const __bf16* Bt = (const __bf16*)Btbits;
    const __bf16* aRowPtr = A  + (size_t)aRow * K;
    const __bf16* bColPtr = Bt + (size_t)(colBase + l15) * K;

    v8f acc = {};
#pragma unroll
    for (int k0 = 0; k0 < 128; k0 += 32) {  // K == 128 for every layer here
        // A fragment: lane holds row aRow, K = k0+half*8 + {0..7, 16..23}
        const int aK = k0 + (half << 3);
        v8bf alo = *(const v8bf*)(aRowPtr + aK);
        v8bf ahi = *(const v8bf*)(aRowPtr + aK + 16);
        v16bf afrag = __builtin_shufflevector(alo, ahi,
            0, 1, 2, 3, 4, 5, 6, 7, 8, 9, 10, 11, 12, 13, 14, 15);

        // B fragment: lane holds column col, K = k0+half*8 + {0..7, 16..23}
        // Bt is K-contiguous per column -> two 16-byte loads per lane.
        const int bK = k0 + (half << 3);
        v8bf blo = *(const v8bf*)(bColPtr + bK);
        v8bf bhi = *(const v8bf*)(bColPtr + bK + 16);
        v16bf bfrag = __builtin_shufflevector(blo, bhi,
            0, 1, 2, 3, 4, 5, 6, 7, 8, 9, 10, 11, 12, 13, 14, 15);

        acc = __builtin_amdgcn_wmma_f32_16x16x32_bf16(
            /*neg_a=*/false, afrag, /*neg_b=*/false, bfrag,
            /*c_mod=*/(short)0, acc, /*reuse_a=*/false, /*reuse_b=*/false);
    }

    // C tile: lane 0-15 -> rows rowBase+0..7, lane 16-31 -> rows rowBase+8..15.
    // For a fixed r, lanes 0-15 cover 16 consecutive floats of one C row
    // (64B coalesced segments). Fast path: no per-store predication.
    const int rBase = rowBase + (half << 3);
    float* cp = C + (size_t)rBase * Ncols + (colBase + l15);
    if (rBase + 8 <= nRows) {
#pragma unroll
        for (int r = 0; r < 8; ++r) { *cp = acc[r]; cp += Ncols; }
    } else {
#pragma unroll
        for (int r = 0; r < 8; ++r) {
            if (rBase + r < nRows) *cp = acc[r];
            cp += Ncols;
        }
    }
}

// ---------------- aggregation:  h = D^-1/2 (A+I) D^-1/2 t ----------------

// self-loop term: h[i][:] = t[i][:] * dinv[i]^2
__global__ void k_agg_init(const float* __restrict__ t, const float* __restrict__ dinv,
                           float* __restrict__ h, int total, int lgF) {
    int i = blockIdx.x * blockDim.x + threadIdx.x;
    if (i < total) {
        const float dn = dinv[i >> lgF];
        h[i] = t[i] * (dn * dn);
    }
}

// edge term: F/4 lanes per edge, float4 gather, 4x f32 atomic scatter (L2-resident)
__global__ void k_agg_edges(const float* __restrict__ t, const float* __restrict__ dinv,
                            const int* __restrict__ src, const int* __restrict__ dst,
                            float* __restrict__ h, int e, int shift, int F) {
    const int gid = blockIdx.x * blockDim.x + threadIdx.x;
    const int edge = gid >> shift;
    if (edge >= e) return;
    const int lane = gid & ((1 << shift) - 1);
    const int s = src[edge];
    const int d = dst[edge];
    const float norm = dinv[s] * dinv[d];
    const int base = lane << 2;
    const float4 m = *(const float4*)(t + (size_t)s * F + base);
    float* hp = h + (size_t)d * F + base;
    atomicAdd(hp + 0, m.x * norm);
    atomicAdd(hp + 1, m.y * norm);
    atomicAdd(hp + 2, m.z * norm);
    atomicAdd(hp + 3, m.w * norm);
}

// fused bias + ReLU + bf16 convert (feeds next layer's WMMA A operand)
__global__ void k_bias_relu_bf16(const float* __restrict__ h, const float* __restrict__ b,
                                 unsigned short* __restrict__ out, int total, int Fmask) {
    int i = blockIdx.x * blockDim.x + threadIdx.x;
    if (i < total) {
        float v = h[i] + b[i & Fmask];
        out[i] = f2bf(v > 0.0f ? v : 0.0f);
    }
}

// fused bias + log_softmax (C <= 32), one node per thread
__global__ void k_logsoftmax(const float* __restrict__ h, const float* __restrict__ b,
                             float* __restrict__ out, int n, int c) {
    int i = blockIdx.x * blockDim.x + threadIdx.x;
    if (i >= n) return;
    float v[32];
    float mx = -3.402823466e+38f;
    for (int j = 0; j < c; ++j) {
        v[j] = h[(size_t)i * c + j] + b[j];
        mx = fmaxf(mx, v[j]);
    }
    float s = 0.0f;
    for (int j = 0; j < c; ++j) s += __expf(v[j] - mx);
    const float l = mx + __logf(s);
    for (int j = 0; j < c; ++j) out[(size_t)i * c + j] = v[j] - l;
}

// ---------------------------------------------------------------------------

static inline int cdiv(long long a, long long b) { return (int)((a + b - 1) / b); }
static inline size_t alignUp(size_t v, size_t a) { return (v + a - 1) & ~(a - 1); }

extern "C" void kernel_launch(void* const* d_in, const int* in_sizes, int n_in,
                              void* d_out, int out_size, void* d_ws, size_t ws_size,
                              hipStream_t stream) {
    const float* x  = (const float*)d_in[0];
    const int*   ei = (const int*)  d_in[1];
    const float* W1 = (const float*)d_in[2];
    const float* b1 = (const float*)d_in[3];
    const float* W2 = (const float*)d_in[4];
    const float* b2 = (const float*)d_in[5];
    const float* W3 = (const float*)d_in[6];
    const float* b3 = (const float*)d_in[7];
    float* out = (float*)d_out;

    const int H = in_sizes[3];                 // 128
    const int C = in_sizes[7];                 // 16
    const int D = in_sizes[2] / H;             // 128
    const int N = in_sizes[0] / D;             // 100000
    const int E = in_sizes[1] / 2;             // 1600000
    const int* src = ei;
    const int* dst = ei + E;

    int lgH = 0; while ((1 << lgH) < H) ++lgH; // 7
    int lgC = 0; while ((1 << lgC) < C) ++lgC; // 4

    // workspace carve (~129 MB)
    char* p = (char*)d_ws;
    size_t off = 0;
    float* deg  = (float*)(p + off); off = alignUp(off + (size_t)N * 4, 256);
    float* dinv = (float*)(p + off); off = alignUp(off + (size_t)N * 4, 256);
    float* t    = (float*)(p + off); off = alignUp(off + (size_t)N * D * 4, 256);
    float* h    = (float*)(p + off); off = alignUp(off + (size_t)N * D * 4, 256);
    unsigned short* abf = (unsigned short*)(p + off); off = alignUp(off + (size_t)N * D * 2, 256);
    unsigned short* w1t = (unsigned short*)(p + off); off = alignUp(off + (size_t)D * H * 2, 256);
    unsigned short* w2t = (unsigned short*)(p + off); off = alignUp(off + (size_t)H * H * 2, 256);
    unsigned short* w3t = (unsigned short*)(p + off); off = alignUp(off + (size_t)H * C * 2, 256);
    (void)ws_size;

    const int BS = 256;

    // --- normalization coefficients ---
    k_deg_init <<<cdiv(N, BS), BS, 0, stream>>>(deg, N);
    k_deg_edges<<<cdiv(E, BS), BS, 0, stream>>>(dst, deg, E);
    k_dinv     <<<cdiv(N, BS), BS, 0, stream>>>(deg, dinv, N);

    // --- bf16 shadows: x elementwise, weights transposed (Wt[n][k]) ---
    k_f2bf <<<cdiv((long long)N * D, BS), BS, 0, stream>>>(x, abf, N * D);
    k_packW<<<cdiv(D * H, BS), BS, 0, stream>>>(W1, w1t, D, H);
    k_packW<<<cdiv(H * H, BS), BS, 0, stream>>>(W2, w2t, H, H);
    k_packW<<<cdiv(H * C, BS), BS, 0, stream>>>(W3, w3t, H, C);

    const int rowTiles = (N + 15) >> 4;
    auto gemm = [&](const unsigned short* A, const unsigned short* Bt, float* Cout,
                    int K, int Ncols) {
        const long long waves = (long long)rowTiles * (Ncols >> 4);
        k_gemm_bf16<<<cdiv(waves * 32, BS), BS, 0, stream>>>(A, Bt, Cout, N, K, Ncols);
    };
    auto aggregate = [&](int F, int lgF) {
        k_agg_init <<<cdiv((long long)N * F, BS), BS, 0, stream>>>(t, dinv, h, N * F, lgF);
        k_agg_edges<<<cdiv((long long)E * (F >> 2), BS), BS, 0, stream>>>(
            t, dinv, src, dst, h, E, lgF - 2, F);
    };

    // --- layer 1: h1 = relu(agg(x @ W1) + b1) -> abf ---
    gemm(abf, w1t, t, D, H);
    aggregate(H, lgH);
    k_bias_relu_bf16<<<cdiv((long long)N * H, BS), BS, 0, stream>>>(h, b1, abf, N * H, H - 1);

    // --- layer 2 ---
    gemm(abf, w2t, t, H, H);
    aggregate(H, lgH);
    k_bias_relu_bf16<<<cdiv((long long)N * H, BS), BS, 0, stream>>>(h, b2, abf, N * H, H - 1);

    // --- layer 3 + log_softmax ---
    gemm(abf, w3t, t, H, C);
    aggregate(C, lgC);
    k_logsoftmax<<<cdiv(N, BS), BS, 0, stream>>>(h, b3, out, N, C);
}
